// SpectrogramMambaBranch_67353677136111
// MI455X (gfx1250) — compile-verified, matrix operand order checked by reading, not run on hardware
//
#include <hip/hip_runtime.h>
#include <hip/hip_bf16.h>

typedef __attribute__((ext_vector_type(16))) _Float16 v16h;
typedef __attribute__((ext_vector_type(8)))  float    v8f;

#define TOK   27040      // 32 * 65 * 13 tokens
#define DM    64
#define DI    128
#define DS    16

// ---------------------------------------------------------------------------
// Stage 1: 3x3 SAME conv (NCHW, H=65, W=13)
// ---------------------------------------------------------------------------
__global__ void k_conv3x3(const float* __restrict__ X, const float* __restrict__ W,
                          float* __restrict__ O, int Cin, int Cout) {
  int i = blockIdx.x * 256 + threadIdx.x;
  int total = 32 * Cout * 845;
  if (i >= total) return;
  int t = i % 13, f = (i / 13) % 65, c = (i / 845) % Cout, bb = i / (845 * Cout);
  float acc = 0.f;
  for (int ic = 0; ic < Cin; ++ic) {
    const float* xp = X + (size_t)(bb * Cin + ic) * 845;
    const float* wp = W + (size_t)(c * Cin + ic) * 9;
#pragma unroll
    for (int kh = 0; kh < 3; ++kh) {
      int ff = f + kh - 1;
      if (ff < 0 || ff >= 65) continue;
#pragma unroll
      for (int kw = 0; kw < 3; ++kw) {
        int tt = t + kw - 1;
        if (tt < 0 || tt >= 13) continue;
        acc += wp[kh * 3 + kw] * xp[ff * 13 + tt];
      }
    }
  }
  O[i] = acc;
}

// per-channel batch stats -> scale/shift
__global__ __launch_bounds__(256)
void k_chstats(const float* __restrict__ X, const float* __restrict__ g,
               const float* __restrict__ b, float* __restrict__ stats, int C) {
  __shared__ float ss[256], sq[256];
  int c = blockIdx.x;
  float s = 0.f, q = 0.f;
  for (int idx = threadIdx.x; idx < 32 * 845; idx += 256) {
    int bb = idx / 845, r = idx % 845;
    float v = X[(size_t)(bb * C + c) * 845 + r];
    s += v; q += v * v;
  }
  ss[threadIdx.x] = s; sq[threadIdx.x] = q;
  __syncthreads();
  for (int o = 128; o; o >>= 1) {
    if ((int)threadIdx.x < o) { ss[threadIdx.x] += ss[threadIdx.x + o]; sq[threadIdx.x] += sq[threadIdx.x + o]; }
    __syncthreads();
  }
  if (threadIdx.x == 0) {
    float mean = ss[0] / 27040.f;
    float var  = sq[0] / 27040.f - mean * mean;
    float sc   = g[c] * rsqrtf(var + 1e-5f);
    stats[2 * c] = sc; stats[2 * c + 1] = b[c] - mean * sc;
  }
}

__global__ void k_bnrelu(float* __restrict__ X, const float* __restrict__ stats,
                         int C, int total) {
  int i = blockIdx.x * 256 + threadIdx.x;
  if (i >= total) return;
  int c = (i / 845) % C;
  X[i] = fmaxf(X[i] * stats[2 * c] + stats[2 * c + 1], 0.f);
}

// (B,C,F,T) -> token-major (m = b*845 + f*13 + t, channel-minor)
__global__ void k_to_tokens(const float* __restrict__ Cc, float* __restrict__ X) {
  int i = blockIdx.x * 256 + threadIdx.x;
  if (i >= TOK * DM) return;
  int c = i & 63, m = i >> 6;
  int bb = m / 845, r = m % 845;
  X[i] = Cc[(size_t)(bb * 64 + c) * 845 + r];
}

// ---------------------------------------------------------------------------
// Weight prep
// ---------------------------------------------------------------------------
__global__ void k_cvt16(const float* __restrict__ s, _Float16* __restrict__ d, int n) {
  int i = blockIdx.x * 256 + threadIdx.x;
  if (i < n) d[i] = (_Float16)s[i];
}
__global__ void k_pad_xproj(const float* __restrict__ s, _Float16* __restrict__ d) {
  int i = blockIdx.x * 256 + threadIdx.x;           // 48 x 128
  if (i >= 48 * 128) return;
  int row = i / 128, col = i % 128;
  d[i] = (_Float16)(row < 36 ? s[row * 128 + col] : 0.f);
}
__global__ void k_aneg(const float* __restrict__ s, float* __restrict__ d, int n) {
  int i = blockIdx.x * 256 + threadIdx.x;
  if (i < n) d[i] = -__expf(s[i]);
}

// ---------------------------------------------------------------------------
// LayerNorm(64) -> f16, one wave per token (wave32 shuffle reduction)
// ---------------------------------------------------------------------------
__global__ __launch_bounds__(128)
void k_layernorm16(const float* __restrict__ X, const float* __restrict__ g,
                   const float* __restrict__ b, _Float16* __restrict__ XH, int T) {
  int lane = threadIdx.x & 31;
  int tok  = blockIdx.x * 4 + (threadIdx.x >> 5);
  if (tok >= T) return;
  const float* x = X + (size_t)tok * 64;
  float a0 = x[lane], a1 = x[lane + 32];
  float s = a0 + a1, q = a0 * a0 + a1 * a1;
#pragma unroll
  for (int o = 16; o; o >>= 1) { s += __shfl_xor(s, o, 32); q += __shfl_xor(q, o, 32); }
  float mean = s * (1.f / 64.f);
  float var  = q * (1.f / 64.f) - mean * mean;
  float rs   = rsqrtf(var + 1e-5f);
  XH[(size_t)tok * 64 + lane]      = (_Float16)((a0 - mean) * rs * g[lane] + b[lane]);
  XH[(size_t)tok * 64 + lane + 32] = (_Float16)((a1 - mean) * rs * g[lane + 32] + b[lane + 32]);
}

// ---------------------------------------------------------------------------
// WMMA GEMM: C[M][N](f32) = act(A[M][K](f16) @ W[N][K]^T(f16) + bias)
// K specialized at compile time (KCH chunks of 32) -> straight-line wmma chain.
// One wave per 16-row M tile; A fragments register-cached across N tiles.
// CDNA5 wave32 fragment layout per ISA 7.12.2 (lane half-K split, 2x b128/lane).
// ---------------------------------------------------------------------------
__device__ inline v16h load_frag(const _Float16* __restrict__ base, int ld,
                                 int rc, int kb, int lane) {
  const int grp = (lane >> 4) & 1;     // K-half select
  const int r   = lane & 15;           // row (A) / col (B)
  const _Float16* p = base + (size_t)(rc + r) * ld + kb + grp * 8;
  v16h f;
#pragma unroll
  for (int v = 0; v < 4; ++v) {
    f[2 * v]         = p[2 * v];
    f[2 * v + 1]     = p[2 * v + 1];
    f[8 + 2 * v]     = p[16 + 2 * v];
    f[8 + 2 * v + 1] = p[16 + 2 * v + 1];
  }
  return f;
}

template <int KCH>
__global__ __launch_bounds__(128)
void k_gemm_wmma(const _Float16* __restrict__ A, const _Float16* __restrict__ W,
                 float* __restrict__ C, _Float16* __restrict__ Ch,
                 const float* __restrict__ bias,
                 int M, int N, int act) {
  constexpr int K = KCH * 32;
  const int lane  = threadIdx.x & 31;
  const int mtile = blockIdx.x * 4 + (threadIdx.x >> 5);
  if (mtile * 16 >= M) return;                    // wave-uniform exit, EXEC stays full
  const int row0 = mtile * 16;
  v16h afrag[KCH];
#pragma unroll
  for (int kc = 0; kc < KCH; ++kc)
    afrag[kc] = load_frag(A, K, row0, kc * 32, lane);
  const int ntiles = N >> 4;
  for (int nt = 0; nt < ntiles; ++nt) {
    const int col0 = nt * 16;
    if (nt + 1 < ntiles)
      __builtin_prefetch(W + (size_t)(col0 + 16) * K, 0, 1);   // global_prefetch_b8
    v8f acc = {};
#pragma unroll
    for (int kc = 0; kc < KCH; ++kc) {
      v16h bfrag = load_frag(W, K, col0, kc * 32, lane);
      acc = __builtin_amdgcn_wmma_f32_16x16x32_f16(
          false, afrag[kc], false, bfrag, (short)0, acc, false, false);
    }
    const int n = col0 + (lane & 15);
    const int g = (lane >> 4) & 1;
    const float bv = bias ? bias[n] : 0.f;
#pragma unroll
    for (int r = 0; r < 8; ++r) {
      int row = row0 + r + 8 * g;
      float v = acc[r] + bv;
      if (act == 1) v = fmaxf(v, 0.f);
      C[(size_t)row * N + n] = v;
      if (Ch) Ch[(size_t)row * N + n] = (_Float16)v;
    }
  }
}

// ---------------------------------------------------------------------------
// Depthwise causal conv(4) + bias + SiLU; reads xc half of XZ, writes f16
// ---------------------------------------------------------------------------
__global__ void k_dwconv_silu(const float* __restrict__ XZ, const float* __restrict__ w,
                              const float* __restrict__ b, _Float16* __restrict__ XCH,
                              int L) {
  int i = blockIdx.x * 256 + threadIdx.x;
  if (i >= TOK * DI) return;
  int m = i >> 7, d = i & 127;
  int l = m % L;
  float acc = b[d];
#pragma unroll
  for (int j = 0; j < 4; ++j) {
    int off = j - 3;
    if (l + off >= 0) acc += w[d * 4 + j] * XZ[(size_t)(m + off) * 256 + d];
  }
  float s = acc / (1.f + __expf(-acc));
  XCH[i] = (_Float16)s;
}

// ---------------------------------------------------------------------------
// Selective scan: one block per sequence, state (128x16) in registers,
// 2 threads per channel d (8 states each), pair-reduce via wave32 shfl_xor.
// Recomputes delta = softplus(dt @ dt_w^T + dt_b) per step.
// ---------------------------------------------------------------------------
__global__ __launch_bounds__(256)
void k_mamba_scan(const float* __restrict__ XZ, const _Float16* __restrict__ XCH,
                  const float* __restrict__ PROJ, const float* __restrict__ dtw,
                  const float* __restrict__ dtb, const float* __restrict__ Aneg,
                  const float* __restrict__ Dp, _Float16* __restrict__ Y, int L) {
  const int tid = threadIdx.x;
  const int d   = tid >> 1;
  const int sh  = (tid & 1) * 8;
  float a[8], h[8];
#pragma unroll
  for (int j = 0; j < 8; ++j) { a[j] = Aneg[d * DS + sh + j]; h[j] = 0.f; }
  const float w0 = dtw[d * 4 + 0], w1 = dtw[d * 4 + 1];
  const float w2 = dtw[d * 4 + 2], w3 = dtw[d * 4 + 3];
  const float db = dtb[d], dpv = Dp[d];
  const size_t m0 = (size_t)blockIdx.x * L;
  for (int l = 0; l < L; ++l) {
    const size_t m = m0 + l;
    const float* pr = PROJ + m * 48;
    float pd = pr[0] * w0 + pr[1] * w1 + pr[2] * w2 + pr[3] * w3 + db;
    float delta = (pd > 20.f) ? pd : log1pf(__expf(pd));
    float x  = (float)XCH[m * DI + d];
    float dx = delta * x;
    float part = 0.f;
#pragma unroll
    for (int j = 0; j < 8; ++j) {
      float dA = __expf(delta * a[j]);
      h[j] = dA * h[j] + dx * pr[4 + sh + j];     // B cols 4..19
      part += h[j] * pr[20 + sh + j];             // C cols 20..35
    }
    float tot = part + __shfl_xor(part, 1, 32);
    if ((tid & 1) == 0) {
      float z  = XZ[m * 256 + DI + d];
      float sz = z / (1.f + __expf(-z));
      Y[m * DI + d] = (_Float16)((tot + x * dpv) * sz);
    }
  }
}

// ---------------------------------------------------------------------------
// Head
// ---------------------------------------------------------------------------
__global__ void k_pool(const float* __restrict__ X2, _Float16* __restrict__ P) {
  int b = blockIdx.x, c = threadIdx.x;            // 32 blocks x 64 threads
  float s = 0.f;
  for (int i = 0; i < 845; ++i) s += X2[((size_t)b * 845 + i) * 64 + c];
  P[b * 64 + c] = (_Float16)(s * (1.f / 845.f));
}

__global__ __launch_bounds__(128)
void k_head_bn_l2(const float* __restrict__ E, const float* __restrict__ g,
                  const float* __restrict__ b, float* __restrict__ out) {
  __shared__ float sm[32 * 128];
  __shared__ float nrm[32];
  int j = threadIdx.x;                            // 128 columns
  float s = 0.f, q = 0.f;
  for (int r = 0; r < 32; ++r) { float v = E[r * 128 + j]; s += v; q += v * v; }
  float mean = s / 32.f, var = q / 32.f - mean * mean;
  float rs = rsqrtf(var + 1e-5f);
  for (int r = 0; r < 32; ++r)
    sm[r * 128 + j] = (E[r * 128 + j] - mean) * rs * g[j] + b[j];
  __syncthreads();
  if (j < 32) {
    float t = 0.f;
    for (int c = 0; c < 128; ++c) { float v = sm[j * 128 + c]; t += v * v; }
    nrm[j] = fmaxf(sqrtf(t), 1e-12f);
  }
  __syncthreads();
  for (int r = 0; r < 32; ++r) out[r * 128 + j] = sm[r * 128 + j] / nrm[r];
}

// ---------------------------------------------------------------------------
// Host
// ---------------------------------------------------------------------------
static inline int cdiv(int a, int b) { return (a + b - 1) / b; }

static void launch_gemm(const _Float16* A, const _Float16* W, float* C, _Float16* Ch,
                        const float* bias, int M, int K, int N, int act,
                        hipStream_t stream) {
  int blocks = cdiv(M / 16, 4);
  if (K == 64)
    k_gemm_wmma<2><<<blocks, 128, 0, stream>>>(A, W, C, Ch, bias, M, N, act);
  else if (K == 128)
    k_gemm_wmma<4><<<blocks, 128, 0, stream>>>(A, W, C, Ch, bias, M, N, act);
  else
    k_gemm_wmma<8><<<blocks, 128, 0, stream>>>(A, W, C, Ch, bias, M, N, act);
}

extern "C" void kernel_launch(void* const* d_in, const int* in_sizes, int n_in,
                              void* d_out, int out_size, void* d_ws, size_t ws_size,
                              hipStream_t stream) {
  const float* x       = (const float*)d_in[0];
  const float* conv1_w = (const float*)d_in[1];
  const float* bn1_g   = (const float*)d_in[2];
  const float* bn1_b   = (const float*)d_in[3];
  const float* conv2_w = (const float*)d_in[4];
  const float* bn2_g   = (const float*)d_in[5];
  const float* bn2_b   = (const float*)d_in[6];
  // fm params 7..17, tm params 18..28
  const float* h1_w  = (const float*)d_in[29];
  const float* h1_b  = (const float*)d_in[30];
  const float* h2_w  = (const float*)d_in[31];
  const float* h2_b  = (const float*)d_in[32];
  const float* hbn_g = (const float*)d_in[33];
  const float* hbn_b = (const float*)d_in[34];

  char* wsp = (char*)d_ws;
  auto take = [&](size_t bytes) {
    char* p = wsp;
    wsp += (bytes + 255) & ~(size_t)255;
    return p;
  };
  float*    C1    = (float*)take(865280u * 4);        // conv1 buf; reused as XH (f16)
  float*    C2    = (float*)take(1730560u * 4);       // conv2 buf; reused as Y16 (f16)
  float*    Xtok  = (float*)take(1730560u * 4);       // X0 / X1 (token-major)
  float*    X2    = (float*)take(1730560u * 4);       // mamba2 output
  float*    XZ    = (float*)take((size_t)TOK * 256 * 4);
  _Float16* XCH   = (_Float16*)take((size_t)TOK * DI * 2);
  float*    PROJ  = (float*)take((size_t)TOK * 48 * 4);
  _Float16* WIN   = (_Float16*)take(16384u * 2);
  _Float16* WXP   = (_Float16*)take(6144u * 2);
  _Float16* WOUT  = (_Float16*)take(8192u * 2);
  _Float16* WH1   = (_Float16*)take(16384u * 2);
  _Float16* WH2   = (_Float16*)take(32768u * 2);
  float*    ANEG  = (float*)take(2048u * 4);
  float*    STATS = (float*)take(128u * 4);
  _Float16* PH    = (_Float16*)take(2048u * 2);
  float*    H1F   = (float*)take(8192u * 4);
  _Float16* H1H   = (_Float16*)take(8192u * 2);
  float*    EMB   = (float*)take(4096u * 4);
  _Float16* XH    = (_Float16*)C1;
  _Float16* Y16   = (_Float16*)C2;

  // ---- CNN front end ----
  k_conv3x3<<<cdiv(865280, 256), 256, 0, stream>>>(x, conv1_w, C1, 4, 32);
  k_chstats<<<32, 256, 0, stream>>>(C1, bn1_g, bn1_b, STATS, 32);
  k_bnrelu<<<cdiv(865280, 256), 256, 0, stream>>>(C1, STATS, 32, 865280);
  k_conv3x3<<<cdiv(1730560, 256), 256, 0, stream>>>(C1, conv2_w, C2, 32, 64);
  k_chstats<<<64, 256, 0, stream>>>(C2, bn2_g, bn2_b, STATS, 64);
  k_bnrelu<<<cdiv(1730560, 256), 256, 0, stream>>>(C2, STATS, 64, 1730560);
  k_to_tokens<<<cdiv(TOK * DM, 256), 256, 0, stream>>>(C2, Xtok);

  // ---- Mamba pass (shared) ----
  auto mamba_pass = [&](int base, const float* Xin, float* Xout, int L) {
    const float* ln_g  = (const float*)d_in[base + 0];
    const float* ln_b  = (const float*)d_in[base + 1];
    const float* in_w  = (const float*)d_in[base + 2];
    const float* cw    = (const float*)d_in[base + 3];
    const float* cb    = (const float*)d_in[base + 4];
    const float* xp_w  = (const float*)d_in[base + 5];
    const float* dt_w  = (const float*)d_in[base + 6];
    const float* dt_b  = (const float*)d_in[base + 7];
    const float* A_log = (const float*)d_in[base + 8];
    const float* Dp    = (const float*)d_in[base + 9];
    const float* out_w = (const float*)d_in[base + 10];

    k_cvt16<<<cdiv(16384, 256), 256, 0, stream>>>(in_w, WIN, 16384);
    k_pad_xproj<<<cdiv(6144, 256), 256, 0, stream>>>(xp_w, WXP);
    k_cvt16<<<cdiv(8192, 256), 256, 0, stream>>>(out_w, WOUT, 8192);
    k_aneg<<<cdiv(2048, 256), 256, 0, stream>>>(A_log, ANEG, 2048);

    k_layernorm16<<<cdiv(TOK, 4), 128, 0, stream>>>(Xin, ln_g, ln_b, XH, TOK);
    launch_gemm(XH, WIN, XZ, nullptr, nullptr, TOK, 64, 256, 0, stream);
    k_dwconv_silu<<<cdiv(TOK * DI, 256), 256, 0, stream>>>(XZ, cw, cb, XCH, L);
    launch_gemm(XCH, WXP, PROJ, nullptr, nullptr, TOK, 128, 48, 0, stream);
    k_mamba_scan<<<TOK / L, 256, 0, stream>>>(XZ, XCH, PROJ, dt_w, dt_b, ANEG, Dp,
                                              Y16, L);
    launch_gemm(Y16, WOUT, Xout, nullptr, nullptr, TOK, 128, 64, 0, stream);
  };

  mamba_pass(7,  Xtok, Xtok, 65);   // frequency-direction mamba (416 seqs x 65)
  mamba_pass(18, Xtok, X2,   13);   // time-direction mamba (2080 seqs x 13)

  // ---- Head ----
  k_cvt16<<<cdiv(16384, 256), 256, 0, stream>>>(h1_w, WH1, 16384);
  k_cvt16<<<cdiv(32768, 256), 256, 0, stream>>>(h2_w, WH2, 32768);
  k_pool<<<32, 64, 0, stream>>>(X2, PH);
  launch_gemm(PH, WH1, H1F, H1H, h1_b, 32, 64, 256, 1, stream);
  launch_gemm(H1H, WH2, EMB, nullptr, h2_b, 32, 256, 128, 0, stream);
  k_head_bn_l2<<<1, 128, 0, stream>>>(EMB, hbn_g, hbn_b, (float*)d_out);

  (void)in_sizes; (void)n_in; (void)out_size; (void)ws_size;
}
